// GCNII_lyc_26826365731122
// MI455X (gfx1250) — matrix operand order repaired and checked by reading, not compiled, not run on hardware
//
#include <hip/hip_runtime.h>
#include <hip/hip_bf16.h>
#include <math.h>

typedef __attribute__((ext_vector_type(2)))  float  v2f;
typedef __attribute__((ext_vector_type(8)))  float  v8f;
typedef __attribute__((ext_vector_type(16))) __bf16 v16bf;

#define KC   32          // original-K slice staged per GEMM iteration
#define TRIP 96          // tripled bf16 K (hi|lo|hi)
#define KPAD 104         // row pad: 104*2B = 208B, 16B-aligned, conflict-free
#define TPB  256         // 8 waves of 32

// ---------------- bf16 split helpers (truncation split: lo = x - hi exact) --
__device__ __forceinline__ unsigned short top16(float f) {
  return (unsigned short)(__builtin_bit_cast(unsigned, f) >> 16);
}
__device__ __forceinline__ float hi_part(float f) {
  return __builtin_bit_cast(float, __builtin_bit_cast(unsigned, f) & 0xFFFF0000u);
}
__device__ __forceinline__ void stage_A3(unsigned short (&As)[128][KPAD],
                                         int r, int c, float v) {
  float h = hi_part(v);
  unsigned short hb = top16(v), lb = top16(v - h);
  As[r][c] = hb; As[r][32 + c] = lb; As[r][64 + c] = hb;
}
__device__ __forceinline__ void stage_B3(unsigned short (&Bs)[64][KPAD],
                                         int n, int c, float v) {
  float h = hi_part(v);
  unsigned short hb = top16(v), lb = top16(v - h);
  Bs[n][c] = hb; Bs[n][32 + c] = hb; Bs[n][64 + c] = lb;
}

// -------- shared bf16x3 inner sweep: 3 segments x (2x2 tiles) = 12 WMMA -----
__device__ __forceinline__ void mm_bf16_slice(const unsigned short (&As)[128][KPAD],
                                              const unsigned short (&Bs)[64][KPAD],
                                              int wm, int wn, int lane, v8f* acc) {
  const int mrow = lane & 15;
  const int kbA = (lane < 16) ? 0 : 8;    // A frag: K {kb..kb+7, 16+kb..16+kb+7}
  const int kbB = (lane < 16) ? 0 : 8;    // B frag: K {kb..kb+7, 16+kb..16+kb+7}
  const int r0 = wm * 32 + mrow, r1 = r0 + 16;
  const int c0 = wn * 32 + mrow, c1 = c0 + 16;   // B stored N-major
  #pragma unroll
  for (int seg = 0; seg < 3; ++seg) {
    const int s = seg * 32;
    v16bf a0, a1, b0, b1;
    #pragma unroll
    for (int t = 0; t < 8; ++t) {
      a0[t]     = __builtin_bit_cast(__bf16, As[r0][s + kbA + t]);
      a0[t + 8] = __builtin_bit_cast(__bf16, As[r0][s + 16 + kbA + t]);
      a1[t]     = __builtin_bit_cast(__bf16, As[r1][s + kbA + t]);
      a1[t + 8] = __builtin_bit_cast(__bf16, As[r1][s + 16 + kbA + t]);
    }
    #pragma unroll
    for (int t = 0; t < 16; ++t) {
      b0[t] = __builtin_bit_cast(__bf16, Bs[c0][s + (t < 8 ? 0 : 8) + kbB + t]);
      b1[t] = __builtin_bit_cast(__bf16, Bs[c1][s + (t < 8 ? 0 : 8) + kbB + t]);
    }
    acc[0] = __builtin_amdgcn_wmma_f32_16x16x32_bf16(false, a0, false, b0, (short)0, acc[0], false, false);
    acc[1] = __builtin_amdgcn_wmma_f32_16x16x32_bf16(false, a0, false, b1, (short)0, acc[1], false, false);
    acc[2] = __builtin_amdgcn_wmma_f32_16x16x32_bf16(false, a1, false, b0, (short)0, acc[2], false, false);
    acc[3] = __builtin_amdgcn_wmma_f32_16x16x32_bf16(false, a1, false, b1, (short)0, acc[3], false, false);
  }
}

// ---------------------------------------------------------------------------
// Kernel 1: row-normalize x -> xn
// ---------------------------------------------------------------------------
__global__ __launch_bounds__(TPB) void rownorm_kernel(const float* __restrict__ x,
                                                      float* __restrict__ xn,
                                                      int F) {
  const int row = blockIdx.x;
  const float* xr = x + (size_t)row * F;
  float ss = 0.f;
  for (int j = threadIdx.x; j < F; j += TPB) { float v = xr[j]; ss += v * v; }
  #pragma unroll
  for (int o = 16; o > 0; o >>= 1) ss += __shfl_down(ss, o, 32);
  __shared__ float red[TPB / 32];
  if ((threadIdx.x & 31) == 0) red[threadIdx.x >> 5] = ss;
  __syncthreads();
  if (threadIdx.x == 0) {
    float t = 0.f;
    #pragma unroll
    for (int w = 0; w < TPB / 32; ++w) t += red[w];
    red[0] = rsqrtf(t);
  }
  __syncthreads();
  const float inv = red[0];
  float* xo = xn + (size_t)row * F;
  for (int j = threadIdx.x; j < F; j += TPB) xo[j] = xr[j] * inv;
}

// ---------------------------------------------------------------------------
// Kernel 2: per-dialogue sim = acos(0.99999 * xn_b @ xn_b^T), full-fp32 WMMA
// (kept in V_WMMA_F32_16X16X4_F32: arccos near 1 is too error-sensitive)
// ---------------------------------------------------------------------------
__global__ __launch_bounds__(TPB) void gram_sim_kernel(const float* __restrict__ xn,
                                                       float* __restrict__ sim,
                                                       int L, int F) {
  const int b = blockIdx.z;
  const float* Xb = xn + (size_t)b * L * F;
  float* Sb = sim + (size_t)b * L * L;
  const int m0 = blockIdx.x * 64;
  const int n0 = blockIdx.y * 32;

  __shared__ float As[64][KC + 1];
  __shared__ float Bs[KC][32 + 1];

  const int wave = threadIdx.x >> 5, lane = threadIdx.x & 31;
  const int wm = wave >> 1, wn = wave & 1;
  const int mrow = lane & 15;
  const int koff = (lane < 16) ? 0 : 2;

  v8f acc = {};
  for (int k0 = 0; k0 < F; k0 += KC) {
    for (int idx = threadIdx.x; idx < 64 * KC; idx += TPB) {
      int r = idx >> 5, c = idx & (KC - 1);
      int gr = m0 + r;
      As[r][c] = (gr < L) ? Xb[(size_t)gr * F + k0 + c] : 0.f;
      if (gr < L && k0 + KC < F)
        __builtin_prefetch(&Xb[(size_t)gr * F + k0 + KC + c], 0, 1);
    }
    for (int idx = threadIdx.x; idx < KC * 32; idx += TPB) {
      int kk = idx >> 5, nn = idx & 31;
      int gn = n0 + nn;
      Bs[kk][nn] = (gn < L) ? Xb[(size_t)gn * F + k0 + kk] : 0.f;
    }
    __syncthreads();
    #pragma unroll
    for (int k = 0; k < KC; k += 4) {
      v2f a, bv;
      a.x  = As[wm * 16 + mrow][k + koff];
      a.y  = As[wm * 16 + mrow][k + koff + 1];
      bv.x = Bs[k + koff][wn * 16 + mrow];
      bv.y = Bs[k + koff + 1][wn * 16 + mrow];
      acc = __builtin_amdgcn_wmma_f32_16x16x4_f32(false, a, false, bv,
                                                  (short)0, acc, false, false);
    }
    __syncthreads();
  }
  const int col = n0 + wn * 16 + mrow;
  const int rbase = m0 + wm * 16 + ((lane < 16) ? 0 : 8);
  if (col < L) {
    #pragma unroll
    for (int v = 0; v < 8; ++v) {
      int row = rbase + v;
      if (row < L) Sb[(size_t)row * L + col] = acosf(0.99999f * acc[v]);
    }
  }
}

// ---------------------------------------------------------------------------
// Kernel 3: dinv[g] = rsqrt(rowsum(sim_row_g))
// ---------------------------------------------------------------------------
__global__ __launch_bounds__(TPB) void rowsum_kernel(const float* __restrict__ sim,
                                                     float* __restrict__ dinv,
                                                     int L) {
  const int g = blockIdx.x;
  const float* Sr = sim + (size_t)g * L;
  float s = 0.f;
  for (int j = threadIdx.x; j < L; j += TPB) s += Sr[j];
  #pragma unroll
  for (int o = 16; o > 0; o >>= 1) s += __shfl_down(s, o, 32);
  __shared__ float red[TPB / 32];
  if ((threadIdx.x & 31) == 0) red[threadIdx.x >> 5] = s;
  __syncthreads();
  if (threadIdx.x == 0) {
    float t = 0.f;
    #pragma unroll
    for (int w = 0; w < TPB / 32; ++w) t += red[w];
    dinv[g] = rsqrtf(t);
  }
}

// ---------------------------------------------------------------------------
// Kernel 4: adj = dinv_i * sim * dinv_j  (in place)
// ---------------------------------------------------------------------------
__global__ __launch_bounds__(TPB) void adjscale_kernel(float* __restrict__ sim,
                                                       const float* __restrict__ dinv,
                                                       int L, int Bn) {
  size_t idx = (size_t)blockIdx.x * TPB + threadIdx.x;
  size_t total = (size_t)Bn * L * L;
  if (idx >= total) return;
  size_t LL = (size_t)L * L;
  size_t b = idx / LL;
  size_t rem = idx - b * LL;
  int i = (int)(rem / L);
  int j = (int)(rem - (size_t)i * L);
  sim[idx] *= dinv[b * L + i] * dinv[b * L + j];
}

// ---------------------------------------------------------------------------
// Kernel 5: h0 = relu(x @ W0 + b0)  — bf16x3, macro 128x64, 2x2 per wave
// ---------------------------------------------------------------------------
__global__ __launch_bounds__(TPB) void h0_gemm_kernel(const float* __restrict__ x,
                                                      const float* __restrict__ W0,
                                                      const float* __restrict__ b0,
                                                      float* __restrict__ h0,
                                                      int F, int H) {
  const int m0 = blockIdx.x * 128;
  const int n0 = blockIdx.y * 64;
  __shared__ __align__(16) unsigned short As[128][KPAD];
  __shared__ __align__(16) unsigned short Bs[64][KPAD];
  const int wave = threadIdx.x >> 5, lane = threadIdx.x & 31;
  const int wm = wave >> 1, wn = wave & 1;

  v8f acc[4] = {};
  for (int k0 = 0; k0 < F; k0 += KC) {
    for (int idx = threadIdx.x; idx < 128 * KC; idx += TPB) {
      int r = idx >> 5, c = idx & (KC - 1);
      stage_A3(As, r, c, x[(size_t)(m0 + r) * F + k0 + c]);
    }
    for (int idx = threadIdx.x; idx < KC * 64; idx += TPB) {
      int kk = idx >> 6, nn = idx & 63;
      stage_B3(Bs, nn, kk, W0[(size_t)(k0 + kk) * H + n0 + nn]);
    }
    __syncthreads();
    mm_bf16_slice(As, Bs, wm, wn, lane, acc);
    __syncthreads();
  }
  const int lane15 = lane & 15;
  #pragma unroll
  for (int ti = 0; ti < 2; ++ti) {
    #pragma unroll
    for (int tj = 0; tj < 2; ++tj) {
      const v8f c = acc[ti * 2 + tj];
      const int col = n0 + wn * 32 + tj * 16 + lane15;
      const int rb  = m0 + wm * 32 + ti * 16 + ((lane < 16) ? 0 : 8);
      const float bias = b0[col];
      #pragma unroll
      for (int v = 0; v < 8; ++v) {
        float t = c[v] + bias;
        h0[(size_t)(rb + v) * H + col] = t > 0.f ? t : 0.f;
      }
    }
  }
}

// ---------------------------------------------------------------------------
// Kernel 6a: hi_b = adj_b @ h_b per dialogue — bf16x3
// ---------------------------------------------------------------------------
__global__ __launch_bounds__(TPB) void prop_gemm_kernel(const float* __restrict__ adj,
                                                        const float* __restrict__ h,
                                                        float* __restrict__ hi,
                                                        int L, int H) {
  const int b = blockIdx.z;
  const float* Ab = adj + (size_t)b * L * L;
  const float* Hb = h + (size_t)b * L * H;
  float* Ob = hi + (size_t)b * L * H;
  const int m0 = blockIdx.x * 128;
  const int n0 = blockIdx.y * 64;
  __shared__ __align__(16) unsigned short As[128][KPAD];
  __shared__ __align__(16) unsigned short Bs[64][KPAD];
  const int wave = threadIdx.x >> 5, lane = threadIdx.x & 31;
  const int wm = wave >> 1, wn = wave & 1;

  v8f acc[4] = {};
  for (int k0 = 0; k0 < L; k0 += KC) {
    for (int idx = threadIdx.x; idx < 128 * KC; idx += TPB) {
      int r = idx >> 5, c = idx & (KC - 1);
      int gr = m0 + r, gk = k0 + c;
      float v = (gr < L && gk < L) ? Ab[(size_t)gr * L + gk] : 0.f;
      stage_A3(As, r, c, v);
    }
    for (int idx = threadIdx.x; idx < KC * 64; idx += TPB) {
      int kk = idx >> 6, nn = idx & 63;
      int gk = k0 + kk;
      float v = (gk < L) ? Hb[(size_t)gk * H + n0 + nn] : 0.f;
      stage_B3(Bs, nn, kk, v);
    }
    __syncthreads();
    mm_bf16_slice(As, Bs, wm, wn, lane, acc);
    __syncthreads();
  }
  const int lane15 = lane & 15;
  #pragma unroll
  for (int ti = 0; ti < 2; ++ti) {
    #pragma unroll
    for (int tj = 0; tj < 2; ++tj) {
      const v8f c = acc[ti * 2 + tj];
      const int col = n0 + wn * 32 + tj * 16 + lane15;
      const int rb  = m0 + wm * 32 + ti * 16 + ((lane < 16) ? 0 : 8);
      #pragma unroll
      for (int v = 0; v < 8; ++v) {
        int row = rb + v;
        if (row < L) Ob[(size_t)row * H + col] = c[v];
      }
    }
  }
}

// ---------------------------------------------------------------------------
// Kernel 6b: h_new = relu(theta*([hi,h0]@Wc_i) + (1-theta)*(0.9*hi+0.1*h0))
// ---------------------------------------------------------------------------
__global__ __launch_bounds__(TPB) void layer_gemm_kernel(const float* __restrict__ hi,
                                                         const float* __restrict__ h0,
                                                         const float* __restrict__ Wci,
                                                         float* __restrict__ hout,
                                                         int H, float theta) {
  const int m0 = blockIdx.x * 128;
  const int n0 = blockIdx.y * 64;
  const int K = 2 * H;
  __shared__ __align__(16) unsigned short As[128][KPAD];
  __shared__ __align__(16) unsigned short Bs[64][KPAD];
  const int wave = threadIdx.x >> 5, lane = threadIdx.x & 31;
  const int wm = wave >> 1, wn = wave & 1;

  v8f acc[4] = {};
  for (int k0 = 0; k0 < K; k0 += KC) {
    for (int idx = threadIdx.x; idx < 128 * KC; idx += TPB) {
      int r = idx >> 5, c = idx & (KC - 1);
      int gk = k0 + c;
      size_t row = (size_t)(m0 + r);
      float v = (gk < H) ? hi[row * H + gk] : h0[row * H + (gk - H)];
      stage_A3(As, r, c, v);
    }
    for (int idx = threadIdx.x; idx < KC * 64; idx += TPB) {
      int kk = idx >> 6, nn = idx & 63;
      stage_B3(Bs, nn, kk, Wci[(size_t)(k0 + kk) * H + n0 + nn]);
    }
    __syncthreads();
    mm_bf16_slice(As, Bs, wm, wn, lane, acc);
    __syncthreads();
  }
  const int lane15 = lane & 15;
  const float omt = 1.f - theta;
  #pragma unroll
  for (int ti = 0; ti < 2; ++ti) {
    #pragma unroll
    for (int tj = 0; tj < 2; ++tj) {
      const v8f c = acc[ti * 2 + tj];
      const int col = n0 + wn * 32 + tj * 16 + lane15;
      const int rb  = m0 + wm * 32 + ti * 16 + ((lane < 16) ? 0 : 8);
      #pragma unroll
      for (int v = 0; v < 8; ++v) {
        size_t off = (size_t)(rb + v) * H + col;
        float r = 0.9f * hi[off] + 0.1f * h0[off];
        float t = theta * c[v] + omt * r;
        hout[off] = t > 0.f ? t : 0.f;
      }
    }
  }
}

// ---------------------------------------------------------------------------
// Kernel 7: out = concat([x, h], axis=-1)
// ---------------------------------------------------------------------------
__global__ __launch_bounds__(TPB) void concat_kernel(const float* __restrict__ x,
                                                     const float* __restrict__ h,
                                                     float* __restrict__ out,
                                                     int F, int H) {
  const int row = blockIdx.x;
  const int OD = F + H;
  for (int j = threadIdx.x; j < OD; j += TPB) {
    float v = (j < F) ? x[(size_t)row * F + j] : h[(size_t)row * H + (j - F)];
    out[(size_t)row * OD + j] = v;
  }
}

// ---------------------------------------------------------------------------
extern "C" void kernel_launch(void* const* d_in, const int* in_sizes, int n_in,
                              void* d_out, int out_size, void* d_ws, size_t ws_size,
                              hipStream_t stream) {
  const float* x  = (const float*)d_in[0];
  const float* W0 = (const float*)d_in[1];
  const float* b0 = (const float*)d_in[2];
  const float* Wc = (const float*)d_in[3];

  const int H  = in_sizes[2];                 // 256
  const int F  = in_sizes[1] / H;             // 1024
  const int N  = in_sizes[0] / F;             // 16000
  const int NL = in_sizes[3] / (2 * H * H);   // 8
  const int L  = 400;                         // structural constant of reference
  const int Bn = N / L;                       // 40

  float* ws = (float*)d_ws;
  size_t off = 0;
  float* xn   = ws + off; off += (size_t)N * F;
  float* sim  = ws + off; off += (size_t)Bn * L * L;
  float* dinv = ws + off; off += (size_t)N;
  float* h0   = ws + off; off += (size_t)N * H;
  float* hA   = ws + off; off += (size_t)N * H;
  float* hB   = ws + off; off += (size_t)N * H;
  float* hi   = ws + off; off += (size_t)N * H;

  const int mt128 = (N + 127) / 128;     // 125
  const int nt64  = (H + 63) / 64;       // 4
  const int lmt64 = (L + 63) / 64;       // 7  (gram M tiles)
  const int lnt32 = (L + 31) / 32;       // 13 (gram N tiles)
  const int lmt128 = (L + 127) / 128;    // 4  (prop M tiles)

  rownorm_kernel<<<N, TPB, 0, stream>>>(x, xn, F);
  gram_sim_kernel<<<dim3(lmt64, lnt32, Bn), TPB, 0, stream>>>(xn, sim, L, F);
  rowsum_kernel<<<N, TPB, 0, stream>>>(sim, dinv, L);
  {
    size_t total = (size_t)Bn * L * L;
    int blocks = (int)((total + TPB - 1) / TPB);
    adjscale_kernel<<<blocks, TPB, 0, stream>>>(sim, dinv, L, Bn);
  }
  h0_gemm_kernel<<<dim3(mt128, nt64), TPB, 0, stream>>>(x, W0, b0, h0, F, H);

  const float* hcur = h0;
  for (int i = 0; i < NL; ++i) {
    float* hnext = (i & 1) ? hB : hA;
    float theta = logf(0.5f / (float)(i + 1) + 1.0f);   // LAMDA = 0.5
    prop_gemm_kernel<<<dim3(lmt128, nt64, Bn), TPB, 0, stream>>>(sim, hcur, hi, L, H);
    layer_gemm_kernel<<<dim3(mt128, nt64), TPB, 0, stream>>>(hi, h0,
                                                             Wc + (size_t)i * 2 * H * H,
                                                             hnext, H, theta);
    hcur = hnext;
  }
  concat_kernel<<<N, TPB, 0, stream>>>(x, hcur, (float*)d_out, F, H);
}
